// WordRNN_3504693313789
// MI455X (gfx1250) — matrix operand order repaired and checked by reading, not compiled, not run on hardware
//
#include <hip/hip_runtime.h>
#include <hip/hip_bf16.h>

typedef __attribute__((ext_vector_type(16))) _Float16 v16h;
typedef __attribute__((ext_vector_type(8)))  float    v8f;

#define TLEN 512
#define MT   4      // 64 rows / 16
#define KT_H 8      // 256 K / 32

__device__ __forceinline__ v8f wmma_acc(v8f c, v16h a, v16h b) {
  // D = A(16x32 f16) * B(32x16 f16) + C(16x16 f32)
  return __builtin_amdgcn_wmma_f32_16x16x32_f16(false, a, false, b, (short)0, c, false, false);
}

__device__ __forceinline__ float sigmoidf_(float x) {
  return 1.0f / (1.0f + __expf(-x));
}

// ---------------------------------------------------------------------------
// Gather embeddings and write them directly in WMMA A-fragment order, f16,
// K padded 50 -> 64.  Layout per t: [mt(4)][kt(2)][lane(32)] x 16 f16.
// A-frag: M = lane&15 (batch row), K = kt*32 + (e&7) + 16*(e>>3) + 8*(lane>>4)
// ---------------------------------------------------------------------------
__global__ void k_embed_pack(const int* __restrict__ x, const float* __restrict__ emb,
                             v16h* __restrict__ xeP) {
  int gid = blockIdx.x * blockDim.x + threadIdx.x;
  if (gid >= TLEN * 4 * 2 * 32) return;
  int lane = gid & 31;
  int kt   = (gid >> 5) & 1;
  int mt   = (gid >> 6) & 3;
  int t    = gid >> 8;
  int b    = mt * 16 + (lane & 15);
  long long tok = x[b * TLEN + t];
  const float* row = emb + tok * 50;
  int kb = kt * 32 + ((lane >> 4) << 3);
  v16h frag;
#pragma unroll
  for (int e = 0; e < 16; ++e) {
    int k = kb + (e & 7) + ((e >> 3) << 4);
    frag[e] = (_Float16)((k < 50) ? row[k] : 0.0f);
  }
  xeP[((t * 4 + mt) * 2 + kt) * 32 + lane] = frag;
}

// ---------------------------------------------------------------------------
// Pack a (768 x Kdim) f32 weight matrix (used as B = W^T) into WMMA B-frag
// order, f16.  Layout: [nt(48)][kt(ktTiles)][lane(32)] x 16 f16.
// B-frag: N = lane&15, K = kt*32 + (e&7) + 16*(e>>3) + 8*(lane>>4)   (zero-pad K >= Kdim)
// ---------------------------------------------------------------------------
__global__ void k_pack_w(const float* __restrict__ W, v16h* __restrict__ dst,
                         int Kdim, int ktTiles) {
  int gid  = blockIdx.x * blockDim.x + threadIdx.x;
  int lane = gid & 31;
  int rest = gid >> 5;
  int kt = rest % ktTiles;
  int nt = rest / ktTiles;
  if (nt >= 48) return;
  int n  = nt * 16 + (lane & 15);
  int kb = kt * 32 + ((lane >> 4) << 3);
  v16h frag;
#pragma unroll
  for (int e = 0; e < 16; ++e) {
    int k = kb + (e & 7) + ((e >> 3) << 4);
    frag[e] = (_Float16)((k < Kdim) ? W[n * Kdim + k] : 0.0f);
  }
  dst[(nt * ktTiles + kt) * 32 + lane] = frag;
}

// ---------------------------------------------------------------------------
// Persistent single-workgroup GRU layer.  512 threads = 16 waves; wave jt owns
// column tile jt (16 cols) of each of the 3 gates across all 4 row tiles.
// All 3 gates accumulate in one fused K loop: per kt, 3 B-frag loads (one
// clause) + 4 A-frag loads feed 12 independent WMMAs (no D->A/B hazards).
// K loops are partially unrolled (x2) to cap the live-register window; the
// h state lives only in LDS (hbuf, lane-owned positions) so peak pressure is
// ~210 VGPRs (128 accum + 48 B-frags + addressing) -> no scratch spills.
// ---------------------------------------------------------------------------
template <int KTIN>
__global__ void __launch_bounds__(512, 1)
k_gru(const v16h* __restrict__ xA,
      const v16h* __restrict__ Wih, const v16h* __restrict__ Whh,
      const float* __restrict__ bih, const float* __restrict__ bhh,
      v16h* __restrict__ hOut, float* __restrict__ pooledOut) {
  extern __shared__ __align__(32) unsigned char smem[];
  float* hbuf  = (float*)smem;                               // 64*256 f32  (64 KB)
  v16h*  hpack = (v16h*)(smem + 65536);                      // 4*8*32 frags (32 KB)
  float* pmax  = (float*)(smem + 65536 + 32768);             // 64*256 f32  (64 KB)
  float* psum  = (float*)(smem + 65536 + 32768 + 65536);     // 64*256 f32  (64 KB)

  const int tid  = threadIdx.x;
  const int lane = tid & 31;
  const int jt   = tid >> 5;        // 0..15 column tile (within a gate)
  const int nl   = lane & 15;
  const int hh   = lane >> 4;       // lane half
  const int j    = jt * 16 + nl;    // hidden column 0..255

  const v16h vh0 = {};
  const v8f  v0  = {};

  for (int i = tid; i < 64 * 256; i += 512) {
    hbuf[i] = 0.0f; pmax[i] = -3.4e38f; psum[i] = 0.0f;
  }
  for (int i = tid; i < 4 * 8 * 32; i += 512) hpack[i] = vh0;
  __syncthreads();

  const float brsum = bih[j] + bhh[j];                 // r gate bias
  const float bzsum = bih[256 + j] + bhh[256 + j];     // z gate bias
  const float bihn  = bih[512 + j];
  const float bhhn  = bhh[512 + j];

  // Per-wave weight fragment bases: gate g tile = (g*16 + jt)
  const v16h* WihR = Wih + (0 * 16 + jt) * (KTIN * 32) + lane;
  const v16h* WihZ = Wih + (1 * 16 + jt) * (KTIN * 32) + lane;
  const v16h* WihN = Wih + (2 * 16 + jt) * (KTIN * 32) + lane;
  const v16h* WhhR = Whh + (0 * 16 + jt) * (KT_H * 32) + lane;
  const v16h* WhhZ = Whh + (1 * 16 + jt) * (KT_H * 32) + lane;
  const v16h* WhhN = Whh + (2 * 16 + jt) * (KT_H * 32) + lane;

  const int aStride = MT * KTIN * 32;

  for (int t = 0; t < TLEN; ++t) {
    const v16h* xAt = xA + (long long)t * aStride + lane;
    if (t + 1 < TLEN)
      __builtin_prefetch((const void*)(xA + (long long)(t + 1) * aStride + lane), 0, 0);

    v8f accR[MT], accZ[MT], accNX[MT], accNH[MT];
#pragma unroll
    for (int mt = 0; mt < MT; ++mt) {
      accR[mt] = v0; accZ[mt] = v0; accNX[mt] = v0; accNH[mt] = v0;
    }

    // ---- input GEMM: all 3 gates share each A fragment ---------------------
#pragma unroll 2
    for (int kt = 0; kt < KTIN; ++kt) {
      v16h br = WihR[kt * 32];
      v16h bz = WihZ[kt * 32];
      v16h bn = WihN[kt * 32];
#pragma unroll
      for (int mt = 0; mt < MT; ++mt) {
        v16h a = xAt[(mt * KTIN + kt) * 32];
        accR[mt]  = wmma_acc(accR[mt],  a, br);
        accZ[mt]  = wmma_acc(accZ[mt],  a, bz);
        accNX[mt] = wmma_acc(accNX[mt], a, bn);
      }
    }
    // ---- hidden GEMM: A fragments from LDS --------------------------------
#pragma unroll 2
    for (int kt = 0; kt < KT_H; ++kt) {
      v16h br = WhhR[kt * 32];
      v16h bz = WhhZ[kt * 32];
      v16h bn = WhhN[kt * 32];
#pragma unroll
      for (int mt = 0; mt < MT; ++mt) {
        v16h a = hpack[(mt * KT_H + kt) * 32 + lane];
        accR[mt]  = wmma_acc(accR[mt],  a, br);
        accZ[mt]  = wmma_acc(accZ[mt],  a, bz);
        accNH[mt] = wmma_acc(accNH[mt], a, bn);
      }
    }

    // ---- gate epilogue: r,z,n + state update + fused pooling ---------------
    // h_old is read from LDS (lane-owned positions) instead of registers.
#pragma unroll
    for (int mt = 0; mt < MT; ++mt) {
      const int mbase = mt * 16 + (hh << 3);   // D layout: M = reg + 8*(lane>>4)
#pragma unroll
      for (int e = 0; e < 8; ++e) {
        int idx = (mbase + e) * 256 + j;
        float hold = hbuf[idx];
        float rv = sigmoidf_(accR[mt][e] + brsum);
        float zv = sigmoidf_(accZ[mt][e] + bzsum);
        float nv = tanhf(accNX[mt][e] + bihn + rv * (accNH[mt][e] + bhhn));
        float hn = (1.0f - zv) * nv + zv * hold;
        hbuf[idx] = hn;
        if (pooledOut) {
          pmax[idx] = fmaxf(pmax[idx], hn);
          psum[idx] += hn;
        }
      }
    }
    __syncthreads();
    // ---- repack new h into A-fragment layout (LDS + optional global) -------
#pragma unroll
    for (int f = jt; f < MT * KT_H; f += 16) {
      int mt = f >> 3, kt = f & 7;
      int m = mt * 16 + nl;
      v16h pk;
#pragma unroll
      for (int e = 0; e < 16; ++e) {
        int k = kt * 32 + (e & 7) + ((e >> 3) << 4) + (hh << 3);
        pk[e] = (_Float16)hbuf[m * 256 + k];
      }
      hpack[(mt * KT_H + kt) * 32 + lane] = pk;
      if (hOut) hOut[(long long)t * (MT * KT_H * 32) + (mt * KT_H + kt) * 32 + lane] = pk;
    }
    __syncthreads();
  }

  if (pooledOut) {
    const float inv = 1.0f / (float)TLEN;
    for (int i = tid; i < 64 * 256; i += 512) {
      int b = i >> 8, c = i & 255;
      pooledOut[b * 512 + c]       = pmax[i];
      pooledOut[b * 512 + 256 + c] = psum[i] * inv;
    }
  }
}

// ---------------------------------------------------------------------------
// Final FC: (64 x 512) @ (5 x 512)^T + b -> (64 x 5)
// ---------------------------------------------------------------------------
__global__ void k_fc(const float* __restrict__ pooled, const float* __restrict__ fcw,
                     const float* __restrict__ fcb, float* __restrict__ out) {
  int i = threadIdx.x;
  if (i >= 320) return;
  int b = i / 5, c = i % 5;
  float acc = fcb[c];
  for (int k = 0; k < 512; ++k) acc += pooled[b * 512 + k] * fcw[c * 512 + k];
  out[b * 5 + c] = acc;
}

extern "C" void kernel_launch(void* const* d_in, const int* in_sizes, int n_in,
                              void* d_out, int out_size, void* d_ws, size_t ws_size,
                              hipStream_t stream) {
  const int*   x    = (const int*)  d_in[0];
  const float* emb  = (const float*)d_in[1];
  const float* Wih0 = (const float*)d_in[2];
  const float* Whh0 = (const float*)d_in[3];
  const float* bih0 = (const float*)d_in[4];
  const float* bhh0 = (const float*)d_in[5];
  const float* Wih1 = (const float*)d_in[6];
  const float* Whh1 = (const float*)d_in[7];
  const float* bih1 = (const float*)d_in[8];
  const float* bhh1 = (const float*)d_in[9];
  const float* fcw  = (const float*)d_in[10];
  const float* fcb  = (const float*)d_in[11];

  unsigned char* ws = (unsigned char*)d_ws;
  size_t off = 0;
  auto carve = [&](size_t bytes) -> void* {
    void* p = ws + off;
    off = (off + bytes + 255) & ~(size_t)255;
    return p;
  };
  v16h*  xeP    = (v16h*)carve(512ull * 4 * 2 * 32 * 32);   // 4 MB  packed embeddings
  v16h*  Wih0p  = (v16h*)carve(48ull * 2 * 32 * 32);        // 96 KB
  v16h*  Whh0p  = (v16h*)carve(48ull * 8 * 32 * 32);        // 384 KB
  v16h*  Wih1p  = (v16h*)carve(48ull * 8 * 32 * 32);        // 384 KB
  v16h*  Whh1p  = (v16h*)carve(48ull * 8 * 32 * 32);        // 384 KB
  v16h*  h1seq  = (v16h*)carve(512ull * 1024 * 32);         // 16 MB layer-0 output, packed
  float* pooled = (float*)carve(64ull * 512 * 4);           // 128 KB

  // parallel pre-pack phase
  k_embed_pack<<<512, 256, 0, stream>>>(x, emb, xeP);
  k_pack_w<<<(48 * 2 * 32 + 255) / 256, 256, 0, stream>>>(Wih0, Wih0p, 50, 2);
  k_pack_w<<<(48 * 8 * 32 + 255) / 256, 256, 0, stream>>>(Whh0, Whh0p, 256, 8);
  k_pack_w<<<(48 * 8 * 32 + 255) / 256, 256, 0, stream>>>(Wih1, Wih1p, 256, 8);
  k_pack_w<<<(48 * 8 * 32 + 255) / 256, 256, 0, stream>>>(Whh1, Whh1p, 256, 8);

  // sequential recurrence: one persistent workgroup per layer
  const size_t smem = 65536 + 32768 + 65536 + 65536;  // 224 KB of the 320 KB WGP LDS
  k_gru<2><<<1, 512, smem, stream>>>(xeP,   Wih0p, Whh0p, bih0, bhh0, h1seq, nullptr);
  k_gru<8><<<1, 512, smem, stream>>>(h1seq, Wih1p, Whh1p, bih1, bhh1, nullptr, pooled);

  k_fc<<<1, 320, 0, stream>>>(pooled, fcw, fcb, (float*)d_out);
}
